// GNNCritic_75625784148324
// MI455X (gfx1250) — matrix operand order repaired and checked by reading, not compiled, not run on hardware
//
#include <hip/hip_runtime.h>
#include <hip/hip_bf16.h>

#define N_NODES  100000
#define N_EDGES  1600000
#define N_GRAPHS 64
#define F_IN  128
#define F_HID 128
#define F_EMB 64
#define F_VAL 64

typedef _Float16 half_t;
typedef __attribute__((ext_vector_type(16))) _Float16 v16h;
typedef __attribute__((ext_vector_type(8)))  float    v8f;
typedef __attribute__((ext_vector_type(4)))  float    v4f;

// ---------------------------------------------------------------- utilities
__global__ void zero_f32_kernel(float* __restrict__ p, int n) {
  int stride = gridDim.x * blockDim.x;
  for (int i = blockIdx.x * blockDim.x + threadIdx.x; i < n; i += stride)
    p[i] = 0.0f;
}

__global__ void cvt_f16_kernel(const float* __restrict__ w,
                               half_t* __restrict__ o, int n) {
  int i = blockIdx.x * blockDim.x + threadIdx.x;
  if (i < n) o[i] = (half_t)w[i];
}

// degree count: one thread per edge
__global__ void deg_kernel(const int* __restrict__ dst, float* __restrict__ deg,
                           int n_edges) {
  int e = blockIdx.x * blockDim.x + threadIdx.x;
  if (e < n_edges) unsafeAtomicAdd(&deg[dst[e]], 1.0f);
}

// in-place deg -> 1/max(deg,1)
__global__ void recip_kernel(float* __restrict__ deg, int n) {
  int i = blockIdx.x * blockDim.x + threadIdx.x;
  if (i < n) deg[i] = 1.0f / fmaxf(deg[i], 1.0f);
}

// ------------------------------------------------- edge scatter (aggregation)
// One wave per edge: 32 lanes x float4 = one full 128-f32 row, coalesced
// gather from h[src], 4 f32 hardware atomics per lane into msg[dst].
// h (51MB) + msg (51MB) + edges (13MB) are L2-resident on MI455X (192MB L2).
__global__ void scatter_add_kernel(const float* __restrict__ h,
                                   const int* __restrict__ src,
                                   const int* __restrict__ dst,
                                   float* __restrict__ msg, int n_edges) {
  int wave = (int)((blockIdx.x * (size_t)blockDim.x + threadIdx.x) >> 5);
  int lane = threadIdx.x & 31;
  if (wave >= n_edges) return;
  int s = src[wave];
  int d = dst[wave];
  v4f v = *((const v4f*)(h + (size_t)s * 128) + lane);
  float* mp = msg + (size_t)d * 128 + lane * 4;
  unsafeAtomicAdd(mp + 0, v[0]);
  unsafeAtomicAdd(mp + 1, v[1]);
  unsafeAtomicAdd(mp + 2, v[2]);
  unsafeAtomicAdd(mp + 3, v[3]);
}

// -------------------------------------------------------------- WMMA GEMM
// out[n][o] = relu?( sum_k msg[n][k]*invdeg[n]*Wl[o][k] + h[n][k]*Wr[o][k] + b[o] )
// One wave -> one 16x16 tile of out; block covers 16 rows x OUT cols.
// K = 128 = 4 WMMA steps of K=32, two v_wmma_f32_16x16x32_f16 per step.
template <int OUT, bool RELU>
__global__ __launch_bounds__(32 * (OUT / 16))
void sage_gemm_kernel(const float* __restrict__ msg,
                      const float* __restrict__ h,
                      const float* __restrict__ invdeg,
                      const half_t* __restrict__ WlH,   // [OUT,128] f16
                      const half_t* __restrict__ WrH,   // [OUT,128] f16
                      const float* __restrict__ bias,   // [OUT]
                      float* __restrict__ out) {        // [N,OUT]
  const int wave    = threadIdx.x >> 5;
  const int lane    = threadIdx.x & 31;
  const int m       = lane & 15;
  const int halfsel = lane >> 4;
  const int row     = blockIdx.x * 16 + m;     // A-matrix row fed by this lane
  const int obase   = wave * 16;

  const float scale = invdeg[row];
  const float* msg_row = msg + (size_t)row * 128;
  const float* h_row   = h   + (size_t)row * 128;
  const half_t* wl_row = WlH + (size_t)(obase + m) * 128;
  const half_t* wr_row = WrH + (size_t)(obase + m) * 128;

  v8f c = {};
#pragma unroll
  for (int kt = 0; kt < 4; ++kt) {
    const int k0 = kt * 32 + halfsel * 16;     // one 16-float run: k0..k0+15
    // A fragment from msg (fold the mean: scale by 1/deg)
    const v4f* mp = (const v4f*)(msg_row + k0);
    v4f m0 = mp[0], m1 = mp[1], m2 = mp[2], m3 = mp[3];
    const v4f* hp = (const v4f*)(h_row + k0);
    v4f h0 = hp[0], h1 = hp[1], h2 = hp[2], h3 = hp[3];
    v16h a_m, a_h;
#pragma unroll
    for (int j = 0; j < 4; ++j) {
      a_m[j]      = (half_t)(m0[j] * scale);
      a_m[4 + j]  = (half_t)(m1[j] * scale);
      a_m[8 + j]  = (half_t)(m2[j] * scale);
      a_m[12 + j] = (half_t)(m3[j] * scale);
      a_h[j]      = (half_t)h0[j];
      a_h[4 + j]  = (half_t)h1[j];
      a_h[8 + j]  = (half_t)h2[j];
      a_h[12 + j] = (half_t)h3[j];
    }
    // B fragments: 16 contiguous halves (32B) per lane from the f16 weights
    const int kb = kt * 32 + halfsel * 16;
    v16h b_l = *(const v16h*)(wl_row + kb);
    v16h b_r = *(const v16h*)(wr_row + kb);
    c = __builtin_amdgcn_wmma_f32_16x16x32_f16(false, a_m, false, b_l,
                                               (short)0, c, false, false);
    c = __builtin_amdgcn_wmma_f32_16x16x32_f16(false, a_h, false, b_r,
                                               (short)0, c, false, false);
  }

  // C/D layout: lane<16 -> (M=r, N=lane); lane>=16 -> (M=8+r, N=lane-16)
  const int ncol  = lane & 15;
  const int mbase = blockIdx.x * 16 + halfsel * 8;
  const float bval = bias[obase + ncol];
#pragma unroll
  for (int r = 0; r < 8; ++r) {
    float v = c[r] + bval;
    if (RELU) v = fmaxf(v, 0.0f);
    out[(size_t)(mbase + r) * OUT + obase + ncol] = v;
  }
}

// ------------------------------------------------------------- graph pooling
// One wave per node: 2 feats/lane into g_sum[batch[n]], + count.
__global__ void pool_kernel(const float* __restrict__ h,    // [N,64]
                            const int* __restrict__ batch,
                            float* __restrict__ g_sum,      // [G,64]
                            float* __restrict__ g_cnt,      // [G]
                            int n_nodes) {
  int wave = (int)((blockIdx.x * (size_t)blockDim.x + threadIdx.x) >> 5);
  int lane = threadIdx.x & 31;
  if (wave >= n_nodes) return;
  int b = batch[wave];
  const float* hp = h + (size_t)wave * 64;
  unsafeAtomicAdd(&g_sum[b * 64 + lane * 2 + 0], hp[lane * 2 + 0]);
  unsafeAtomicAdd(&g_sum[b * 64 + lane * 2 + 1], hp[lane * 2 + 1]);
  if (lane == 0) unsafeAtomicAdd(&g_cnt[b], 1.0f);
}

// ----------------------------------------------------------------- MLP head
// 64 graphs x tiny MLP -> single workgroup, LDS-staged.
__global__ __launch_bounds__(256)
void head_kernel(const float* __restrict__ g_sum, const float* __restrict__ g_cnt,
                 const float* __restrict__ Wv1, const float* __restrict__ bv1,
                 const float* __restrict__ Wv2, const float* __restrict__ bv2,
                 const float* __restrict__ Wv3, const float* __restrict__ bv3,
                 float* __restrict__ out) {
  __shared__ float buf_a[N_GRAPHS * 64];
  __shared__ float buf_b[N_GRAPHS * 64];
  const int t = threadIdx.x;
  for (int i = t; i < N_GRAPHS * 64; i += 256) {
    int g = i >> 6;
    buf_a[i] = g_sum[i] * (1.0f / fmaxf(g_cnt[g], 1.0f));
  }
  __syncthreads();
  for (int i = t; i < N_GRAPHS * 64; i += 256) {
    int g = i >> 6, o = i & 63;
    float acc = bv1[o];
    for (int k = 0; k < 64; ++k) acc += buf_a[g * 64 + k] * Wv1[o * 64 + k];
    buf_b[i] = fmaxf(acc, 0.0f);
  }
  __syncthreads();
  for (int i = t; i < N_GRAPHS * 64; i += 256) {
    int g = i >> 6, o = i & 63;
    float acc = bv2[o];
    for (int k = 0; k < 64; ++k) acc += buf_b[g * 64 + k] * Wv2[o * 64 + k];
    buf_a[i] = fmaxf(acc, 0.0f);
  }
  __syncthreads();
  if (t < N_GRAPHS) {
    float acc = bv3[0];
    for (int k = 0; k < 64; ++k) acc += buf_a[t * 64 + k] * Wv3[k];
    out[t] = acc;
  }
}

// ------------------------------------------------------------------- driver
extern "C" void kernel_launch(void* const* d_in, const int* in_sizes, int n_in,
                              void* d_out, int out_size, void* d_ws, size_t ws_size,
                              hipStream_t stream) {
  (void)in_sizes; (void)n_in; (void)out_size; (void)ws_size;
  const float* x     = (const float*)d_in[0];
  const int*   ei    = (const int*)d_in[1];       // [2, E] (jax int64 -> int32)
  const int*   batch = (const int*)d_in[2];
  const float* Wl0 = (const float*)d_in[3];
  const float* Wr0 = (const float*)d_in[4];
  const float* b0  = (const float*)d_in[5];
  const float* Wl1 = (const float*)d_in[6];
  const float* Wr1 = (const float*)d_in[7];
  const float* b1  = (const float*)d_in[8];
  const float* Wl2 = (const float*)d_in[9];
  const float* Wr2 = (const float*)d_in[10];
  const float* b2  = (const float*)d_in[11];
  const float* Wv1 = (const float*)d_in[12];
  const float* bv1 = (const float*)d_in[13];
  const float* Wv2 = (const float*)d_in[14];
  const float* bv2 = (const float*)d_in[15];
  const float* Wv3 = (const float*)d_in[16];
  const float* bv3 = (const float*)d_in[17];
  const int* src  = ei;
  const int* dstE = ei + N_EDGES;
  float* out = (float*)d_out;

  // workspace carve-up (256B aligned)
  char* ws = (char*)d_ws;
  size_t cur = 0;
  auto alloc = [&](size_t bytes) -> char* {
    cur = (cur + 255) & ~(size_t)255;
    char* p = ws + cur;
    cur += bytes;
    return p;
  };
  const size_t big = (size_t)N_NODES * 128 * sizeof(float);
  float* msg   = (float*)alloc(big);
  float* bufA  = (float*)alloc(big);
  float* bufB  = (float*)alloc(big);
  float* deg   = (float*)alloc((size_t)N_NODES * sizeof(float));
  half_t* Wl0h = (half_t*)alloc(128 * 128 * sizeof(half_t));
  half_t* Wr0h = (half_t*)alloc(128 * 128 * sizeof(half_t));
  half_t* Wl1h = (half_t*)alloc(128 * 128 * sizeof(half_t));
  half_t* Wr1h = (half_t*)alloc(128 * 128 * sizeof(half_t));
  half_t* Wl2h = (half_t*)alloc(64 * 128 * sizeof(half_t));
  half_t* Wr2h = (half_t*)alloc(64 * 128 * sizeof(half_t));
  float* g_sum = (float*)alloc(N_GRAPHS * 64 * sizeof(float));
  float* g_cnt = (float*)alloc(N_GRAPHS * sizeof(float));

  const int nElems = N_NODES * 128;
  const int zgrid = 2048, zthr = 256;
  const int row_tiles = N_NODES / 16;                 // 6250 exact
  const int scat_blocks = (N_EDGES + 7) / 8;          // 8 waves/block
  const int pool_blocks = (N_NODES + 7) / 8;

  // degree (shared by all 3 layers), then invert in place
  zero_f32_kernel<<<zgrid, zthr, 0, stream>>>(deg, N_NODES);
  deg_kernel<<<(N_EDGES + 255) / 256, 256, 0, stream>>>(dstE, deg, N_EDGES);
  recip_kernel<<<(N_NODES + 255) / 256, 256, 0, stream>>>(deg, N_NODES);

  // f32 -> f16 weights (one-shot, tiny)
  cvt_f16_kernel<<<(128 * 128 + 255) / 256, 256, 0, stream>>>(Wl0, Wl0h, 128 * 128);
  cvt_f16_kernel<<<(128 * 128 + 255) / 256, 256, 0, stream>>>(Wr0, Wr0h, 128 * 128);
  cvt_f16_kernel<<<(128 * 128 + 255) / 256, 256, 0, stream>>>(Wl1, Wl1h, 128 * 128);
  cvt_f16_kernel<<<(128 * 128 + 255) / 256, 256, 0, stream>>>(Wr1, Wr1h, 128 * 128);
  cvt_f16_kernel<<<(64 * 128 + 255) / 256, 256, 0, stream>>>(Wl2, Wl2h, 64 * 128);
  cvt_f16_kernel<<<(64 * 128 + 255) / 256, 256, 0, stream>>>(Wr2, Wr2h, 64 * 128);

  // ---- layer 0: x -> bufA (relu)
  zero_f32_kernel<<<zgrid, zthr, 0, stream>>>(msg, nElems);
  scatter_add_kernel<<<scat_blocks, 256, 0, stream>>>(x, src, dstE, msg, N_EDGES);
  sage_gemm_kernel<128, true><<<row_tiles, 256, 0, stream>>>(
      msg, x, deg, Wl0h, Wr0h, b0, bufA);

  // ---- layer 1: bufA -> bufB (relu)
  zero_f32_kernel<<<zgrid, zthr, 0, stream>>>(msg, nElems);
  scatter_add_kernel<<<scat_blocks, 256, 0, stream>>>(bufA, src, dstE, msg, N_EDGES);
  sage_gemm_kernel<128, true><<<row_tiles, 256, 0, stream>>>(
      msg, bufA, deg, Wl1h, Wr1h, b1, bufB);

  // ---- layer 2: bufB -> bufA as dense [N,64] (no relu)
  zero_f32_kernel<<<zgrid, zthr, 0, stream>>>(msg, nElems);
  scatter_add_kernel<<<scat_blocks, 256, 0, stream>>>(bufB, src, dstE, msg, N_EDGES);
  sage_gemm_kernel<64, false><<<row_tiles, 128, 0, stream>>>(
      msg, bufB, deg, Wl2h, Wr2h, b2, bufA);

  // ---- global mean pool + value head
  zero_f32_kernel<<<64, 256, 0, stream>>>(g_sum, N_GRAPHS * 64);
  zero_f32_kernel<<<1, 64, 0, stream>>>(g_cnt, N_GRAPHS);
  pool_kernel<<<pool_blocks, 256, 0, stream>>>(bufA, batch, g_sum, g_cnt, N_NODES);
  head_kernel<<<1, 256, 0, stream>>>(g_sum, g_cnt, Wv1, bv1, Wv2, bv2, Wv3, bv3, out);
}
